// QKVBlock_8495445311945
// MI455X (gfx1250) — compile-verified, matrix operand order checked by reading, not run on hardware
//
#include <hip/hip_runtime.h>

// ---------------- problem constants ----------------
#define EMB   1024
#define NH    16
#define HD    64
#define BATCH 2
#define SEQ   2048
#define MTOT  (BATCH*SEQ)   // 4096 rows

typedef _Float16 h4  __attribute__((ext_vector_type(4)));
typedef _Float16 h8  __attribute__((ext_vector_type(8)));
typedef _Float16 v16h __attribute__((ext_vector_type(16)));
typedef float    v8f  __attribute__((ext_vector_type(8)));

static __device__ __forceinline__ v16h make16(h8 lo, h8 hi) {
  return __builtin_shufflevector(lo, hi, 0,1,2,3,4,5,6,7,8,9,10,11,12,13,14,15);
}

static __device__ __forceinline__ v8f wmma_f16(v16h a, v16h b, v8f c) {
  return __builtin_amdgcn_wmma_f32_16x16x32_f16(false, a, false, b, (short)0, c,
                                                false, false);
}

// ---------------- fp32 -> f16 convert ----------------
__global__ void cvt_f32_f16(const float* __restrict__ src,
                            _Float16* __restrict__ dst, int n) {
  int i = (blockIdx.x * blockDim.x + threadIdx.x) * 4;
  if (i < n) {
    float4 v = *(const float4*)(src + i);
    h4 o = { (_Float16)v.x, (_Float16)v.y, (_Float16)v.z, (_Float16)v.w };
    *(h4*)(dst + i) = o;
  }
}

// ---------------- WMMA GEMM:  C[m,n] = sum_k A[m,k]*Bt[n,k] + bias[n] ----------------
// Block: 256 threads = 8 waves (4 along M x 2 along N); block tile 128x128.
// Wave tile: 32(M) x 64(N) = 2x4 WMMA tiles -> 8 WMMAs per 12 b128 loads per K-step.
// MODE 0: fp32 row-major [M,N] to out_f (+bias, scale)
// MODE 1: f16 QKV head layout:
//         m -> (b = m/SEQ, s = m%SEQ), n -> (h = n/HD, d = n%HD)
//         addr = (b*NH+h)*SEQ*HD + s*ss + d*sd   (ss=HD,sd=1 row-major; ss=1,sd=SEQ transposed)
template <int MODE>
__global__ __launch_bounds__(256)
void gemm_f16_wmma(const _Float16* __restrict__ A,
                   const _Float16* __restrict__ Bt,
                   const float* __restrict__ bias,
                   _Float16* __restrict__ out_h,
                   float* __restrict__ out_f,
                   int N, int K, float scale, int ss, int sd) {
  const int lane = threadIdx.x & 31;
  const int wave = threadIdx.x >> 5;
  const int wm = wave & 3;                       // 4 waves along M
  const int wn = wave >> 2;                      // 2 waves along N
  const int m0 = blockIdx.y * 128 + wm * 32;     // 32 rows per wave
  const int n0 = blockIdx.x * 128 + wn * 64;     // 64 cols per wave
  const int lrow = lane & 15;
  const int kb   = (lane >> 4) * 8;              // 16-bit A/B operand lane layout

  const _Float16* aBase = A  + (size_t)(m0 + lrow) * K + kb;
  const _Float16* bBase = Bt + (size_t)(n0 + lrow) * K + kb;

  v8f acc[2][4] = {};

  for (int kk = 0; kk < K; kk += 32) {
    if (kk + 64 < K) {                           // prefetch two K-steps ahead
      __builtin_prefetch(aBase + kk + 64, 0, 0);
      __builtin_prefetch(bBase + kk + 64, 0, 0);
    }
    v16h a[2], b[4];
#pragma unroll
    for (int i = 0; i < 2; ++i) {
      const _Float16* ap = aBase + (size_t)(i * 16) * K + kk;
      a[i] = make16(*(const h8*)ap, *(const h8*)(ap + 16));
    }
#pragma unroll
    for (int j = 0; j < 4; ++j) {
      const _Float16* bp = bBase + (size_t)(j * 16) * K + kk;
      b[j] = make16(*(const h8*)bp, *(const h8*)(bp + 16));
    }
#pragma unroll
    for (int i = 0; i < 2; ++i)
#pragma unroll
      for (int j = 0; j < 4; ++j)
        acc[i][j] = wmma_f16(a[i], b[j], acc[i][j]);
  }

#pragma unroll
  for (int i = 0; i < 2; ++i) {
#pragma unroll
    for (int j = 0; j < 4; ++j) {
      const int col = n0 + j * 16 + lrow;
      const float bv = bias[col];
      const int rbase = m0 + i * 16 + (lane >> 4) * 8;   // C layout rows
#pragma unroll
      for (int r = 0; r < 8; ++r) {
        const int m = rbase + r;
        const float v = (acc[i][j][r] + bv) * scale;
        if (MODE == 0) {
          out_f[(size_t)m * N + col] = v;
        } else {
          const int bb = m >> 11, s = m & (SEQ - 1);
          const int h  = col >> 6, d = col & (HD - 1);
          out_h[(size_t)(bb * NH + h) * (SEQ * HD) +
                (size_t)s * ss + (size_t)d * sd] = (_Float16)v;
        }
      }
    }
  }
}

// ---------------- Flash attention (per wave: 16 q rows, KV tile = 64 keys) ----------------
// Qh,Kh: [B,H,S,HD] f16 (Q pre-scaled).  Vt: [B,H,HD,S] f16.  Oh: [B,S,D] f16.
__global__ __launch_bounds__(128)
void attn_wmma(const _Float16* __restrict__ Qh,
               const _Float16* __restrict__ Kh,
               const _Float16* __restrict__ Vt,
               _Float16* __restrict__ Oh) {
  __shared__ _Float16 ldsP[4][16 * 64];   // per-wave 16x64 prob tile (C-layout -> A-layout)

  const int lane = threadIdx.x & 31;
  const int wave = threadIdx.x >> 5;
  const int bh = blockIdx.y;                 // b*NH + h
  const int b  = bh >> 4;
  const int h  = bh & (NH - 1);
  const int qbase = blockIdx.x * 64 + wave * 16;
  const int lrow = lane & 15;
  const int kb   = (lane >> 4) * 8;
  const size_t qkbase = (size_t)bh * SEQ * HD;

  // Q A-operands: 16 rows x 64 cols = two k-chunks of 32
  v16h aQ[2];
#pragma unroll
  for (int c = 0; c < 2; ++c) {
    const _Float16* p = Qh + qkbase + (size_t)(qbase + lrow) * HD + c * 32 + kb;
    aQ[c] = make16(*(const h8*)p, *(const h8*)(p + 16));
  }

  float mrow[8], lsum[8];
  v8f accO[4] = {};
#pragma unroll
  for (int r = 0; r < 8; ++r) { mrow[r] = -INFINITY; lsum[r] = 0.0f; }

  _Float16* myP = ldsP[wave];
  const size_t vbase = (size_t)bh * HD * SEQ;

  for (int kv = 0; kv < SEQ; kv += 64) {
    // ---- scores: four 16x16 tiles over 64 keys (2 chained WMMAs each) ----
    v8f sc[4];
#pragma unroll
    for (int t = 0; t < 4; ++t) {
      const _Float16* kp = Kh + qkbase + (size_t)(kv + t * 16 + lrow) * HD + kb;
      v16h bk0 = make16(*(const h8*)kp,        *(const h8*)(kp + 16));
      v16h bk1 = make16(*(const h8*)(kp + 32), *(const h8*)(kp + 48));
      v8f z = {};
      z = wmma_f16(aQ[0], bk0, z);
      z = wmma_f16(aQ[1], bk1, z);
      sc[t] = z;
    }

    // ---- online softmax: one reduction tree per row per 64 keys ----
#pragma unroll
    for (int r = 0; r < 8; ++r) {
      float v = fmaxf(fmaxf(sc[0][r], sc[1][r]), fmaxf(sc[2][r], sc[3][r]));
#pragma unroll
      for (int msk = 1; msk < 16; msk <<= 1) v = fmaxf(v, __shfl_xor(v, msk));
      const float mnew = fmaxf(mrow[r], v);
      const float f = __expf(mrow[r] - mnew);
      mrow[r] = mnew;

      float p[4], rs = 0.0f;
#pragma unroll
      for (int t = 0; t < 4; ++t) { p[t] = __expf(sc[t][r] - mnew); rs += p[t]; }
#pragma unroll
      for (int msk = 1; msk < 16; msk <<= 1) rs += __shfl_xor(rs, msk);
      lsum[r] = lsum[r] * f + rs;
#pragma unroll
      for (int t = 0; t < 4; ++t) accO[t][r] *= f;

      const int m = r + (lane >> 4) * 8;     // C-layout row
#pragma unroll
      for (int t = 0; t < 4; ++t)
        myP[m * 64 + t * 16 + lrow] = (_Float16)p[t];
    }

    // wave-private LDS: wait for the ds stores, then re-read in A-layout
    asm volatile("s_wait_dscnt 0" ::: "memory");
    v16h aP[2];
#pragma unroll
    for (int c = 0; c < 2; ++c) {
      const _Float16* p = myP + lrow * 64 + c * 32 + kb;
      aP[c] = make16(*(const h8*)p, *(const h8*)(p + 16));
    }

    // ---- P @ V : 4 hd tiles x 2 k-chunks of 32 keys (Vt rows contiguous in key) ----
#pragma unroll
    for (int t = 0; t < 4; ++t) {
      const _Float16* vp = Vt + vbase + (size_t)(t * 16 + lrow) * SEQ + kv + kb;
      v16h bv0 = make16(*(const h8*)vp,        *(const h8*)(vp + 16));
      v16h bv1 = make16(*(const h8*)(vp + 32), *(const h8*)(vp + 48));
      accO[t] = wmma_f16(aP[0], bv0, accO[t]);
      accO[t] = wmma_f16(aP[1], bv1, accO[t]);
    }
  }

  // ---- normalize and store to [B,S,D] f16 ----
#pragma unroll
  for (int t = 0; t < 4; ++t) {
#pragma unroll
    for (int r = 0; r < 8; ++r) {
      const int m = qbase + r + (lane >> 4) * 8;
      const float v = accO[t][r] / lsum[r];
      Oh[(size_t)(b * SEQ + m) * EMB + h * HD + t * 16 + lrow] = (_Float16)v;
    }
  }
}

// ---------------- host launch ----------------
extern "C" void kernel_launch(void* const* d_in, const int* in_sizes, int n_in,
                              void* d_out, int out_size, void* d_ws, size_t ws_size,
                              hipStream_t stream) {
  (void)in_sizes; (void)n_in; (void)out_size; (void)ws_size;
  const float* x  = (const float*)d_in[0];
  const float* Wq = (const float*)d_in[1];
  const float* bq = (const float*)d_in[2];
  const float* Wk = (const float*)d_in[3];
  const float* bk = (const float*)d_in[4];
  const float* Wv = (const float*)d_in[5];
  const float* bv = (const float*)d_in[6];
  const float* Wo = (const float*)d_in[7];
  const float* bo = (const float*)d_in[8];
  float* out = (float*)d_out;

  char* ws = (char*)d_ws;
  size_t off = 0;
  _Float16* xh  = (_Float16*)(ws + off); off += (size_t)MTOT * EMB * 2;
  _Float16* Wqh = (_Float16*)(ws + off); off += (size_t)EMB * EMB * 2;
  _Float16* Wkh = (_Float16*)(ws + off); off += (size_t)EMB * EMB * 2;
  _Float16* Wvh = (_Float16*)(ws + off); off += (size_t)EMB * EMB * 2;
  _Float16* Woh = (_Float16*)(ws + off); off += (size_t)EMB * EMB * 2;
  _Float16* Qh  = (_Float16*)(ws + off); off += (size_t)MTOT * EMB * 2;
  _Float16* Kh  = (_Float16*)(ws + off); off += (size_t)MTOT * EMB * 2;
  _Float16* Vth = (_Float16*)(ws + off); off += (size_t)MTOT * EMB * 2;
  _Float16* Ah  = (_Float16*)(ws + off); off += (size_t)MTOT * EMB * 2;

  // fp32 -> f16 conversions
  {
    const int nx = MTOT * EMB;
    cvt_f32_f16<<<(nx / 4 + 255) / 256, 256, 0, stream>>>(x, xh, nx);
    const int nw = EMB * EMB;
    const int g  = (nw / 4 + 255) / 256;
    cvt_f32_f16<<<g, 256, 0, stream>>>(Wq, Wqh, nw);
    cvt_f32_f16<<<g, 256, 0, stream>>>(Wk, Wkh, nw);
    cvt_f32_f16<<<g, 256, 0, stream>>>(Wv, Wvh, nw);
    cvt_f32_f16<<<g, 256, 0, stream>>>(Wo, Woh, nw);
  }

  // QKV projections (f16 out, head layouts); q pre-scaled by Hd^-0.5
  dim3 gb(EMB / 128, MTOT / 128);
  const float qscale = 0.125f;   // 64^-0.5
  gemm_f16_wmma<1><<<gb, 256, 0, stream>>>(xh, Wqh, bq, Qh,  nullptr, EMB, EMB,
                                           qscale, HD, 1);
  gemm_f16_wmma<1><<<gb, 256, 0, stream>>>(xh, Wkh, bk, Kh,  nullptr, EMB, EMB,
                                           1.0f, HD, 1);
  gemm_f16_wmma<1><<<gb, 256, 0, stream>>>(xh, Wvh, bv, Vth, nullptr, EMB, EMB,
                                           1.0f, 1, SEQ);   // V transposed [B,H,HD,S]

  // flash attention
  attn_wmma<<<dim3(SEQ / 64, BATCH * NH), 128, 0, stream>>>(Qh, Kh, Vth, Ah);

  // output projection -> fp32 d_out
  gemm_f16_wmma<0><<<gb, 256, 0, stream>>>(Ah, Woh, bo, nullptr, out, EMB, EMB,
                                           1.0f, 0, 0);
}
